// ProjectionScale_28630251995467
// MI455X (gfx1250) — matrix operand order repaired and checked by reading, not compiled, not run on hardware
//
#include <hip/hip_runtime.h>

typedef __attribute__((ext_vector_type(16))) _Float16 v16h;
typedef __attribute__((ext_vector_type(8)))  _Float16 v8h;
typedef __attribute__((ext_vector_type(8)))  float    v8f;

#define B_  2
#define C_  64
#define D_  64
#define H_  128
#define W_  128
#define NX_ 129
#define NZ_ 64
#define AP  136   // padded LDS row stride for A (f16 elems): 272B -> rotates banks

// ---------------------------------------------------------------- zero fill
__global__ void ProjScale_zero_kernel(float* __restrict__ p, int n) {
    int i = blockIdx.x * blockDim.x + threadIdx.x;
    if (i < n) p[i] = 0.0f;
}

// ------------------------------------------------- softmax over D + abs_depth
__global__ void ProjScale_softmax_kernel(const float* __restrict__ logit,
                                         _Float16* __restrict__ prob,
                                         float* __restrict__ abs_depth) {
    int idx = blockIdx.x * blockDim.x + threadIdx.x;
    if (idx >= B_ * H_ * W_) return;
    int b   = idx / (H_ * W_);
    int pix = idx % (H_ * W_);
    const float* lp = logit + (size_t)b * D_ * H_ * W_ + pix;
    _Float16*    qp = prob  + (size_t)b * D_ * H_ * W_ + pix;

    float m = -3.4e38f;
    for (int d = 0; d < D_; ++d) m = fmaxf(m, lp[d * H_ * W_]);
    float s = 0.0f;
    for (int d = 0; d < D_; ++d) s += __expf(lp[d * H_ * W_] - m);
    float inv = 1.0f / s;
    float ad = 0.0f;
    for (int d = 0; d < D_; ++d) {
        float pr = __expf(lp[d * H_ * W_] - m) * inv;
        qp[d * H_ * W_] = (_Float16)pr;
        ad += pr * (0.5f * (float)(d + 1));
    }
    abs_depth[idx] = ad;
}

// --------------------------------------- per-(b,u) 64x64xK128 WMMA GEMM + scatter
__global__ __launch_bounds__(128) void ProjScale_bev_gemm_kernel(
    const _Float16* __restrict__ prob,        // [B][D][H][W] f16
    const float*    __restrict__ image,       // [B][C][H][W]
    const float*    __restrict__ valid_image, // [B][1][H][W]
    const float*    __restrict__ camera,      // [B][4] fx fy cx cy
    float* __restrict__ image_bev,            // [B][C][NX][NZ]
    float* __restrict__ valid_bev,            // [B][1][NX][NZ]
    float* __restrict__ depth_bev)            // [B][1][NX][NZ]
{
    __shared__ __align__(16) _Float16 ldsA[D_ * AP];          // masked prob [d][v]
    // B in WMMA fragment-major layout: [kc][half][c][e], e = 0..15 contiguous.
    // Fragment element e of (kc, half, c) holds image^T[v = kc*32 + half*16 + e][c].
    __shared__ __align__(16) _Float16 ldsBf[4 * 2 * C_ * 16];
    __shared__ float vld[H_];
    __shared__ float rowsum[D_];
    __shared__ int   ixs[D_];
    __shared__ int   okv[D_];
    __shared__ int   nvs[D_];

    const int wg = blockIdx.x;            // b*W + u
    const int b  = wg / W_;
    const int u  = wg % W_;
    const int t  = threadIdx.x;

    const float fx = camera[b * 4 + 0];
    const float fy = camera[b * 4 + 1];
    const float cx = camera[b * 4 + 2];
    const float cy = camera[b * 4 + 3];

    // stage valid_image column (per v, fixed u)
    vld[t & 127] = valid_image[((size_t)b * H_ + (t & 127)) * W_ + u];
    __syncthreads();

    // stage A: masked depth_prob [d][v]
    for (int i = t; i < D_ * H_; i += 128) {
        int d = i >> 7, v = i & 127;
        float z  = 0.5f * (float)(d + 1);
        float pr = (float)prob[(((size_t)b * D_ + d) * H_ + v) * W_ + u];
        float y  = z * ((float)v - cy) / fy;
        if (fabsf(y) > 32.0f || !(vld[v] > 0.0f)) pr = 0.0f;
        ldsA[d * AP + v] = (_Float16)pr;
    }
    // stage B (image^T) directly in fragment-major layout
    for (int i = t; i < 4 * 2 * C_ * 16; i += 128) {
        int e  = i & 15;
        int c  = (i >> 4) & 63;
        int h  = (i >> 10) & 1;
        int kc = i >> 11;
        int v  = kc * 32 + h * 16 + e;
        ldsBf[i] = (_Float16)image[(((size_t)b * C_ + c) * H_ + v) * W_ + u];
    }
    __syncthreads();

    // per-d metadata (waves 0-1 only; uniform per wave, no intra-wave divergence)
    if (t < D_) {
        int d = t;
        float z  = 0.5f * (float)(d + 1);
        float x  = z * ((float)u - cx) / fx;
        int   ix = (int)floorf((x + 32.0f) / 0.5f);
        int   iz = d + 1;
        ixs[d] = ix;
        okv[d] = (ix >= 0 && ix < NX_ && iz >= 0 && iz < NZ_) ? 1 : 0;
        float rs = 0.0f; int nv = 0;
        for (int v = 0; v < H_; ++v) {
            rs += (float)ldsA[d * AP + v];
            float y = z * ((float)v - cy) / fy;
            nv += ((fabsf(y) <= 32.0f) && (vld[v] > 0.0f)) ? 1 : 0;
        }
        rowsum[d] = rs; nvs[d] = nv;
    }

    // ---- GEMM: each wave owns d rows [wave*16, wave*16+16), all 64 c columns
    const int lane   = t & 31;
    const int wave   = t >> 5;
    const int half   = lane >> 4;
    const int lr     = lane & 15;
    const int d_base = wave * 16;

    v8f acc0 = {}, acc1 = {}, acc2 = {}, acc3 = {};
    #pragma unroll
    for (int kc = 0; kc < 4; ++kc) {
        const int k0 = kc * 32;
        // A fragment (16x32 f16): lane holds row d_base+lr; K = {k0+16h..+15}
        v8h a_lo = *(const v8h*)&ldsA[(d_base + lr) * AP + k0 + 16 * half];
        v8h a_hi = *(const v8h*)&ldsA[(d_base + lr) * AP + k0 + 16 * half + 8];
        v16h a = __builtin_shufflevector(a_lo, a_hi,
                                         0, 1, 2, 3, 4, 5, 6, 7,
                                         8, 9, 10, 11, 12, 13, 14, 15);
        // B fragments: two aligned b128 loads per tile from fragment-major LDS
        const _Float16* bp = &ldsBf[((kc * 2 + half) * C_) * 16];
        v8h q0, q1;
        q0 = *(const v8h*)&bp[(lr +  0) * 16];
        q1 = *(const v8h*)&bp[(lr +  0) * 16 + 8];
        v16h bt0 = __builtin_shufflevector(q0, q1, 0,1,2,3,4,5,6,7,8,9,10,11,12,13,14,15);
        q0 = *(const v8h*)&bp[(lr + 16) * 16];
        q1 = *(const v8h*)&bp[(lr + 16) * 16 + 8];
        v16h bt1 = __builtin_shufflevector(q0, q1, 0,1,2,3,4,5,6,7,8,9,10,11,12,13,14,15);
        q0 = *(const v8h*)&bp[(lr + 32) * 16];
        q1 = *(const v8h*)&bp[(lr + 32) * 16 + 8];
        v16h bt2 = __builtin_shufflevector(q0, q1, 0,1,2,3,4,5,6,7,8,9,10,11,12,13,14,15);
        q0 = *(const v8h*)&bp[(lr + 48) * 16];
        q1 = *(const v8h*)&bp[(lr + 48) * 16 + 8];
        v16h bt3 = __builtin_shufflevector(q0, q1, 0,1,2,3,4,5,6,7,8,9,10,11,12,13,14,15);

        acc0 = __builtin_amdgcn_wmma_f32_16x16x32_f16(false, a, false, bt0, (short)0, acc0, false, false);
        acc1 = __builtin_amdgcn_wmma_f32_16x16x32_f16(false, a, false, bt1, (short)0, acc1, false, false);
        acc2 = __builtin_amdgcn_wmma_f32_16x16x32_f16(false, a, false, bt2, (short)0, acc2, false, false);
        acc3 = __builtin_amdgcn_wmma_f32_16x16x32_f16(false, a, false, bt3, (short)0, acc3, false, false);
    }
    __syncthreads();

    // ---- scatter image_bev: D element r -> (d = d_base + r + 8*half, c = tile*16 + lr)
    #pragma unroll
    for (int r = 0; r < 8; ++r) {
        int d = d_base + r + 8 * half;
        if (!okv[d]) continue;
        size_t base = (((size_t)b * C_) * NX_ + ixs[d]) * (size_t)NZ_ + (size_t)(d + 1);
        size_t cs   = (size_t)NX_ * NZ_;
        atomicAdd(&image_bev[base + (size_t)(lr +  0) * cs], acc0[r]);
        atomicAdd(&image_bev[base + (size_t)(lr + 16) * cs], acc1[r]);
        atomicAdd(&image_bev[base + (size_t)(lr + 32) * cs], acc2[r]);
        atomicAdd(&image_bev[base + (size_t)(lr + 48) * cs], acc3[r]);
    }

    // ---- scatter depth_bev / valid_bev (one thread per d)
    if (t < D_) {
        int d = t;
        if (okv[d] && nvs[d] > 0) {
            size_t o = ((size_t)b * NX_ + ixs[d]) * (size_t)NZ_ + (size_t)(d + 1);
            valid_bev[o] = 1.0f;  // idempotent store; races benign
            atomicAdd(&depth_bev[o], (0.5f * (float)(d + 1)) * rowsum[d]);
        }
    }
}

extern "C" void kernel_launch(void* const* d_in, const int* in_sizes, int n_in,
                              void* d_out, int out_size, void* d_ws, size_t ws_size,
                              hipStream_t stream) {
    const float* image       = (const float*)d_in[0];
    const float* valid_image = (const float*)d_in[1];
    const float* depth_logit = (const float*)d_in[2];
    const float* camera      = (const float*)d_in[3];

    float* out       = (float*)d_out;
    float* image_bev = out;                                      // [B][C][NX][NZ]
    float* valid_bev = image_bev + (size_t)B_ * C_ * NX_ * NZ_;  // [B][1][NX][NZ]
    float* abs_depth = valid_bev + (size_t)B_ * NX_ * NZ_;       // [B][1][H][W]
    float* depth_bev = abs_depth + (size_t)B_ * H_ * W_;         // [B][1][NX][NZ]

    _Float16* prob = (_Float16*)d_ws;                            // [B][D][H][W] f16

    ProjScale_zero_kernel<<<(out_size + 255) / 256, 256, 0, stream>>>(out, out_size);

    int npix = B_ * H_ * W_;
    ProjScale_softmax_kernel<<<(npix + 255) / 256, 256, 0, stream>>>(depth_logit, prob, abs_depth);

    ProjScale_bev_gemm_kernel<<<B_ * W_, 128, 0, stream>>>(
        prob, image, valid_image, camera, image_bev, valid_bev, depth_bev);
}
